// NMT_25185688224727
// MI455X (gfx1250) — compile-verified, matrix operand order checked by reading, not run on hardware
//
#include <hip/hip_runtime.h>
#include <math.h>

// ---------------- problem constants ----------------
constexpr int cB = 32, cS = 48, cT = 48;
constexpr int cE = 512, cHH = 512, cH = 1024, cV = 32000;

typedef __bf16 bf16_t;
typedef __attribute__((ext_vector_type(16))) __bf16 v16bf;
typedef __attribute__((ext_vector_type(8)))  float  v8f;
typedef unsigned int v4u __attribute__((ext_vector_type(4)));
typedef int v8i __attribute__((ext_vector_type(8)));
typedef int v4i __attribute__((ext_vector_type(4)));

// ---------------- WMMA helpers (wave32, gfx1250) ----------------
// 16-bit A-matrix 16x32 layout (ISA 7.12.2): lane = (half<<4)|row ;
// VGPR r holds K pair k0 = (r>>2)*16 + half*8 + (r&3)*2.
// B (32x16, from row-major W[n][k]) uses the mirrored layout with row=n.
__device__ inline v16bf load_frag(const bf16_t* __restrict__ base, int stride) {
  int lane = threadIdx.x & 31;
  int row = lane & 15;
  int half = lane >> 4;
  const bf16_t* p = base + (long)row * stride;
  v16bf f;
#pragma unroll
  for (int r = 0; r < 8; ++r) {
    int k = ((r >> 2) << 4) + (half << 3) + ((r & 3) << 1);
    f[2 * r] = p[k];
    f[2 * r + 1] = p[k + 1];
  }
  return f;
}

__device__ inline v8f wmma_bf(v16bf a, v16bf b, v8f c) {
  return __builtin_amdgcn_wmma_f32_16x16x32_bf16(false, a, false, b, (short)0, c,
                                                 false, false);
}

__device__ inline float sigmoidf_(float x) { return 1.0f / (1.0f + __expf(-x)); }

// ---------------- Tensor Data Mover (TDM) helpers ----------------
#if __has_builtin(__builtin_amdgcn_tensor_load_to_lds)
#define HAVE_TDM 1
#endif

#ifdef HAVE_TDM
// DMA a 2D tile (16 rows x 1024 bf16 elements, row stride 2048 elements) from
// global memory into LDS (packed 16x1024). Descriptor per ISA 08 §8.3/8.4.
__device__ inline void tdm_load_tile(const bf16_t* gsrc, unsigned lds_off) {
  unsigned long long ga = (unsigned long long)(uintptr_t)gsrc;
  v4u g0;
  g0.x = 1u;                                    // count=1, user mode
  g0.y = lds_off;                               // lds_addr (bytes)
  g0.z = (unsigned)ga;                          // global_addr[31:0]
  g0.w = (unsigned)(ga >> 32) | (2u << 30);     // global_addr[56:32] | type=2
  v8i g1;
  g1[0] = 0x00010000;                           // wg_mask=0, data_size=2B
  g1[1] = (int)(1024u << 16);                   // tensor_dim0 = 1024
  g1[2] = (int)(16u << 16);                     // tensor_dim1 = 16
  g1[3] = (int)(1024u << 16);                   // tile_dim0 = 1024
  g1[4] = 16;                                   // tile_dim1 = 16, tile_dim2 = 0
  g1[5] = 2048;                                 // tensor_dim0_stride = 2048
  g1[6] = 0;
  g1[7] = 0;
#if __clang_major__ >= 23
  __builtin_amdgcn_tensor_load_to_lds(g0, g1, (v4i)0, (v4i)0, (v8i)0, 0);
#else
  __builtin_amdgcn_tensor_load_to_lds(g0, g1, (v4i)0, (v4i)0, 0);
#endif
}
#endif

// ---------------- global barrier (persistent kernels) ----------------
__device__ inline void gbar(unsigned* cnt, unsigned* gen, unsigned nb) {
  __syncthreads();
  if (threadIdx.x == 0) {
    __threadfence();
    unsigned g = atomicAdd(gen, 0u);
    if (atomicAdd(cnt, 1u) == nb - 1u) {
      atomicExch(cnt, 0u);
      __threadfence();
      atomicAdd(gen, 1u);
    } else {
      while (atomicAdd(gen, 0u) == g) __builtin_amdgcn_s_sleep(2);
    }
  }
  __syncthreads();
}

// ---------------- utility kernels ----------------
__global__ void k_cvt_bf16(const float* __restrict__ s, bf16_t* __restrict__ d, long n) {
  long i = (long)blockIdx.x * blockDim.x + threadIdx.x;
  long st = (long)gridDim.x * blockDim.x;
  for (; i < n; i += st) d[i] = (bf16_t)s[i];
}

__global__ void k_embed_enc(const float* __restrict__ et, const int* __restrict__ tok,
                            const int* __restrict__ slen, bf16_t* __restrict__ eF,
                            bf16_t* __restrict__ eR) {
  int i = blockIdx.x * blockDim.x + threadIdx.x;
  if (i >= cB * cS * cE) return;
  int e = i % cE, bs = i / cE, b = bs / cS, s = bs % cS;
  int len = slen[b];
  int tf = tok[b * cS + s];
  int rs = len - 1 - s; if (rs < 0) rs = 0;
  int tr = tok[b * cS + rs];
  eF[i] = (bf16_t)et[(long)tf * cE + e];
  eR[i] = (bf16_t)et[(long)tr * cE + e];
}

__global__ void k_embed_dec(const float* __restrict__ et, const int* __restrict__ tok,
                            bf16_t* __restrict__ de) {
  int i = blockIdx.x * blockDim.x + threadIdx.x;
  if (i >= cB * cT * cE) return;
  int e = i % cE, bt = i / cE, b = bt / cT, t = bt % cT;
  int tk = tok[b * (cT + 1) + t];  // tgt_in = tgt_tokens[:, :-1]
  de[i] = (bf16_t)et[(long)tk * cE + e];
}

// OUT[M,N] = A[M,K](bf16) @ W[N,K](bf16)^T + b1[n] + b2[n]   (one 16x16 tile / wave)
__global__ void __launch_bounds__(256) k_gemm_bias(
    const bf16_t* __restrict__ A, const bf16_t* __restrict__ W, int ldw,
    const float* __restrict__ b1, const float* __restrict__ b2,
    float* __restrict__ O, int M, int N, int K) {
  int lane = threadIdx.x & 31;
  int gw = blockIdx.x * (blockDim.x >> 5) + (threadIdx.x >> 5);
  int nw = gridDim.x * (blockDim.x >> 5);
  int tm = M >> 4, tn = N >> 4;
  int half = lane >> 4, nn = lane & 15;
  for (int tile = gw; tile < tm * tn; tile += nw) {
    int mt = tile / tn, nt = tile % tn;
    v8f acc = {};
    for (int k = 0; k < K; k += 32) {
      v16bf a = load_frag(A + (long)(mt * 16) * K + k, K);
      v16bf b = load_frag(W + (long)(nt * 16) * ldw + k, ldw);
      acc = wmma_bf(a, b, acc);
    }
    int n = nt * 16 + nn;
    float bv = 0.f;
    if (b1) bv += b1[n];
    if (b2) bv += b2[n];
#pragma unroll
    for (int v = 0; v < 8; ++v) {
      int m = mt * 16 + v + 8 * half;
      O[(long)m * N + n] = acc[v] + bv;
    }
  }
}

// ---------------- persistent bi-directional encoder LSTM ----------------
__global__ void __launch_bounds__(256) k_enc_lstm(
    const float* __restrict__ XF, const float* __restrict__ XB,
    const bf16_t* __restrict__ WhhF, const bf16_t* __restrict__ WhhB,
    const int* __restrict__ slen,
    bf16_t* hFbf, bf16_t* hBbf,          // [2][B*HH] double-buffered
    float* hF, float* cF, float* hB, float* cB2,
    float* outF, float* outR,
    unsigned* cnt, unsigned* gen) {
  const unsigned NB = 16;
  int lane = threadIdx.x & 31;
  int gwave = blockIdx.x * 8 + (threadIdx.x >> 5);
  // zero-init states (buffer 0 only)
  {
    int tid = blockIdx.x * 256 + threadIdx.x;
    for (int i = tid; i < cB * cHH; i += (int)NB * 256) {
      hF[i] = 0.f; cF[i] = 0.f; hB[i] = 0.f; cB2[i] = 0.f;
      hFbf[i] = (bf16_t)0.f; hBbf[i] = (bf16_t)0.f;
    }
  }
  gbar(cnt, gen, NB);

  int dir = gwave >> 6;           // 0 = fwd, 1 = bwd (64 waves each)
  int wid = gwave & 63;
  int j0 = (wid & 31) * 16;       // hidden-col tile (0..511)
  int b0 = (wid >> 5) * 16;       // batch tile (0 or 16)
  const float* X = dir ? XB : XF;
  const bf16_t* Whh = dir ? WhhB : WhhF;
  bf16_t* hbf = dir ? hBbf : hFbf;
  float* hf = dir ? hB : hF;
  float* cf = dir ? cB2 : cF;
  float* out = dir ? outR : outF;
  int half = lane >> 4, nn = lane & 15;
  int j = j0 + nn;

  for (int t = 0; t < cS; ++t) {
    int buf = t & 1, nbuf = buf ^ 1;
    v8f ai = {}, af = {}, ag = {}, ao = {};
    const bf16_t* hsrc = hbf + buf * cB * cHH + b0 * cHH;
    for (int k = 0; k < cHH; k += 32) {
      v16bf a = load_frag(hsrc + k, cHH);
      ai = wmma_bf(a, load_frag(Whh + (long)(0 * cHH + j0) * cHH + k, cHH), ai);
      af = wmma_bf(a, load_frag(Whh + (long)(1 * cHH + j0) * cHH + k, cHH), af);
      ag = wmma_bf(a, load_frag(Whh + (long)(2 * cHH + j0) * cHH + k, cHH), ag);
      ao = wmma_bf(a, load_frag(Whh + (long)(3 * cHH + j0) * cHH + k, cHH), ao);
    }
#pragma unroll
    for (int v = 0; v < 8; ++v) {
      int b = b0 + v + 8 * half;
      long xr = ((long)b * cS + t) * (4 * cHH);
      float gi = ai[v] + X[xr + 0 * cHH + j];
      float gf = af[v] + X[xr + 1 * cHH + j];
      float gg = ag[v] + X[xr + 2 * cHH + j];
      float go = ao[v] + X[xr + 3 * cHH + j];
      long hi = (long)b * cHH + j;
      float c_old = cf[hi], h_old = hf[hi];
      float cn = sigmoidf_(gf) * c_old + sigmoidf_(gi) * tanhf(gg);
      float hn = sigmoidf_(go) * tanhf(cn);
      bool m = t < slen[b];
      float hcur = m ? hn : h_old;
      float ccur = m ? cn : c_old;
      hf[hi] = hcur; cf[hi] = ccur;
      hbf[nbuf * cB * cHH + hi] = (bf16_t)hcur;
      out[((long)b * cS + t) * cHH + j] = m ? hn : 0.f;
    }
    gbar(cnt, gen, NB);
  }
}

// ---------------- post-encoder: src_enc assembly ----------------
__global__ void k_postenc(const float* __restrict__ outF, const float* __restrict__ outR,
                          const int* __restrict__ slen, float* __restrict__ senc) {
  int i = blockIdx.x * blockDim.x + threadIdx.x;
  if (i >= cB * cS * cH) return;
  int hc = i % cH, bs = i / cH, b = bs / cS, s = bs % cS;
  int len = slen[b];
  float v;
  if (hc < cHH) {
    v = outF[((long)b * cS + s) * cHH + hc];
  } else {
    int rs = len - 1 - s; if (rs < 0) rs = 0;
    v = (s < len) ? outR[((long)b * cS + rs) * cHH + (hc - cHH)] : 0.f;
  }
  senc[i] = v;
}

__global__ void k_initdec(const float* __restrict__ hF, const float* __restrict__ hB,
                          const float* __restrict__ cF, const float* __restrict__ cB2,
                          float* __restrict__ hd, float* __restrict__ cd,
                          bf16_t* __restrict__ habf0, bf16_t* __restrict__ wabf0) {
  int i = blockIdx.x * blockDim.x + threadIdx.x;
  if (i >= cB * cH) return;
  int b = i / cH, j = i % cH;
  float hv = (j < cHH) ? hF[b * cHH + j] : hB[b * cHH + j - cHH];
  float cv = (j < cHH) ? cF[b * cHH + j] : cB2[b * cHH + j - cHH];
  hd[i] = hv; cd[i] = cv;
  habf0[i] = (bf16_t)hv;
  wabf0[i] = (bf16_t)0.f;
}

// ---------------- persistent decoder LSTM + attention ----------------
__global__ void __launch_bounds__(256) k_dec_lstm(
    const float* __restrict__ PDE, const bf16_t* __restrict__ Wih,
    const bf16_t* __restrict__ Whh, const float* __restrict__ senc,
    const int* __restrict__ slen,
    bf16_t* habf,   // [2][B*H]
    bf16_t* wabf,   // [2][B*H]
    float* hd, float* cd, bf16_t* OUTS,
    unsigned* cnt, unsigned* gen) {
  const unsigned NB = 16;
  constexpr int LDW = cE + cH;  // 1536
  int lane = threadIdx.x & 31;
  int wix = threadIdx.x >> 5;
  int gwave = blockIdx.x * 8 + wix;
  int j0 = (gwave & 63) * 16;   // hidden col tile (0..1023)
  int b0 = (gwave >> 6) * 16;   // batch tile
  int half = lane >> 4, nn = lane & 15;
  int j = j0 + nn;
  __shared__ float s_align[cS];
  __shared__ float s_a[cS];

  for (int t = 0; t < cT; ++t) {
    int buf = t & 1, nbuf = buf ^ 1;
    v8f ai = {}, af = {}, ag = {}, ao = {};
    const bf16_t* wasrc = wabf + buf * cB * cH + b0 * cH;
    for (int k = 0; k < cH; k += 32) {  // wa @ Wih[:, :H]^T
      v16bf a = load_frag(wasrc + k, cH);
      ai = wmma_bf(a, load_frag(Wih + (long)(0 * cH + j0) * LDW + k, LDW), ai);
      af = wmma_bf(a, load_frag(Wih + (long)(1 * cH + j0) * LDW + k, LDW), af);
      ag = wmma_bf(a, load_frag(Wih + (long)(2 * cH + j0) * LDW + k, LDW), ag);
      ao = wmma_bf(a, load_frag(Wih + (long)(3 * cH + j0) * LDW + k, LDW), ao);
    }
    const bf16_t* hsrc = habf + buf * cB * cH + b0 * cH;
    for (int k = 0; k < cH; k += 32) {  // h @ Whh^T
      v16bf a = load_frag(hsrc + k, cH);
      ai = wmma_bf(a, load_frag(Whh + (long)(0 * cH + j0) * cH + k, cH), ai);
      af = wmma_bf(a, load_frag(Whh + (long)(1 * cH + j0) * cH + k, cH), af);
      ag = wmma_bf(a, load_frag(Whh + (long)(2 * cH + j0) * cH + k, cH), ag);
      ao = wmma_bf(a, load_frag(Whh + (long)(3 * cH + j0) * cH + k, cH), ao);
    }
#pragma unroll
    for (int v = 0; v < 8; ++v) {
      int b = b0 + v + 8 * half;
      long pr = ((long)b * cT + t) * (4 * cH);
      float gi = ai[v] + PDE[pr + 0 * cH + j];
      float gf = af[v] + PDE[pr + 1 * cH + j];
      float gg = ag[v] + PDE[pr + 2 * cH + j];
      float go = ao[v] + PDE[pr + 3 * cH + j];
      long hi = (long)b * cH + j;
      float cn = sigmoidf_(gf) * cd[hi] + sigmoidf_(gi) * tanhf(gg);
      float hn = sigmoidf_(go) * tanhf(cn);
      cd[hi] = cn; hd[hi] = hn;
      habf[nbuf * cB * cH + hi] = (bf16_t)hn;
      OUTS[((long)b * cT + t) * (2 * cH) + cH + j] = (bf16_t)hn;  // out_t = [wa, h]
    }
    gbar(cnt, gen, NB);

    // attention: block handles batches 2*blk and 2*blk+1
    for (int q = 0; q < 2; ++q) {
      int bb = blockIdx.x * 2 + q;
      for (int s = wix; s < cS; s += 8) {
        float p = 0.f;
        const float* se = senc + ((long)bb * cS + s) * cH;
        const float* hh = hd + (long)bb * cH;
        for (int i2 = lane; i2 < cH; i2 += 32) p += se[i2] * hh[i2];
#pragma unroll
        for (int o = 16; o > 0; o >>= 1) p += __shfl_xor(p, o, 32);
        if (lane == 0) s_align[s] = p;
      }
      __syncthreads();
      if (threadIdx.x == 0) {
        int len = slen[bb];
        float mx = -3e38f;
        for (int s = 0; s < len; ++s) mx = fmaxf(mx, s_align[s]);
        float sm = 0.f;
        for (int s = 0; s < len; ++s) sm += __expf(s_align[s] - mx);
        float inv = 1.f / sm;
        for (int s = 0; s < cS; ++s)
          s_a[s] = (s < len) ? __expf(s_align[s] - mx) * inv : 0.f;
      }
      __syncthreads();
      for (int r = 0; r < cH / 256; ++r) {
        int hc = threadIdx.x + 256 * r;
        float acc2 = 0.f;
        for (int s = 0; s < cS; ++s)
          acc2 += s_a[s] * senc[((long)bb * cS + s) * cH + hc];
        wabf[nbuf * cB * cH + (long)bb * cH + hc] = (bf16_t)acc2;
        OUTS[((long)bb * cT + t) * (2 * cH) + hc] = (bf16_t)acc2;
      }
      __syncthreads();
    }
    gbar(cnt, gen, NB);
  }
}

// ---------------- fused vocab GEMM + online log-sum-exp ----------------
// grid = 12 row-groups (128 rows) x 16 vocab slices. 8 waves = 8 m-tiles, all
// sharing one B tile staged in LDS by the Tensor Data Mover (double-buffered
// 16x1024 half-K chunks, overlapped with compute via TENSORcnt).
__global__ void __launch_bounds__(256) k_vocab(
    const bf16_t* __restrict__ OUTS, const bf16_t* __restrict__ Wsb16,
    const float* __restrict__ Wsb, const int* __restrict__ tgt_tokens,
    float* __restrict__ Pmax, float* __restrict__ Psum, float* __restrict__ Ptgt) {
  __shared__ bf16_t Bs[2][16 * 1024];  // 64 KB double-buffered B tile
  int lane = threadIdx.x & 31;
  int wix = threadIdx.x >> 5;
  int rg = blockIdx.x >> 4;       // 0..11
  int slice = blockIdx.x & 15;    // 0..15
  int r0 = rg * 128 + wix * 16;
  int half = lane >> 4, nn = lane & 15;
  constexpr int NT = (cV / 16) / 16;  // 125 vocab tiles per slice
  constexpr int NC = NT * 2;          // 250 half-K chunks
  const v8f vzero = {};

  int tgt[8]; float lmax[8], lsum[8], ltg[8];
#pragma unroll
  for (int v = 0; v < 8; ++v) {
    int row = r0 + v + 8 * half;
    int b = row / cT, t = row % cT;
    tgt[v] = tgt_tokens[b * (cT + 1) + t + 1];
    lmax[v] = -3e38f; lsum[v] = 0.f; ltg[v] = 0.f;
  }

#ifdef HAVE_TDM
  // prologue: issue chunk 0 (wave 0 only; TDM ignores EXEC, one op per wave)
  if (wix == 0) {
    tdm_load_tile(Wsb16 + (long)(slice * 16) * 2048,
                  (unsigned)(uintptr_t)&Bs[0][0]);
  }
#endif

  v8f acc = vzero;
  for (int c = 0; c < NC; ++c) {
    int cti = c >> 1, kh = c & 1, buf = c & 1;
    int ct = slice + cti * 16;
#ifdef HAVE_TDM
    if (wix == 0) {
      int c1 = c + 1;
      if (c1 < NC) {
        int ct1 = slice + (c1 >> 1) * 16;
        int kh1 = c1 & 1;
        tdm_load_tile(Wsb16 + (long)(ct1 * 16) * 2048 + kh1 * 1024,
                      (unsigned)(uintptr_t)&Bs[c1 & 1][0]);
        __builtin_amdgcn_s_wait_tensorcnt(1);  // chunk c resident, c+1 in flight
      } else {
        __builtin_amdgcn_s_wait_tensorcnt(0);
      }
    }
    __syncthreads();
#else
    // fallback: cooperative global->LDS copy of chunk c
    {
      const bf16_t* src = Wsb16 + (long)(ct * 16) * 2048 + kh * 1024;
      for (int i = threadIdx.x; i < 16 * 256; i += 256) {
        int rown = i >> 8, col = i & 255;
        ((unsigned long long*)&Bs[buf][0])[i] =
            ((const unsigned long long*)(src + (long)rown * 2048))[col];
      }
    }
    __syncthreads();
#endif
    // 16x16x1024 half-chunk GEMM: A from global (L1-resident), B from LDS
    const bf16_t* Ab = OUTS + (long)r0 * (2 * cH) + kh * 1024;
    for (int kk = 0; kk < 1024; kk += 32) {
      v16bf a = load_frag(Ab + kk, 2 * cH);
      v16bf b;
#pragma unroll
      for (int r = 0; r < 8; ++r) {
        int kx = ((r >> 2) << 4) + (half << 3) + ((r & 3) << 1) + kk;
        b[2 * r] = Bs[buf][nn * 1024 + kx];
        b[2 * r + 1] = Bs[buf][nn * 1024 + kx + 1];
      }
      acc = wmma_bf(a, b, acc);
    }
    if (kh == 1) {  // full K accumulated for tile ct: online log-sum-exp
      int n = ct * 16 + nn;
      float bv = Wsb[n];
#pragma unroll
      for (int v = 0; v < 8; ++v) {
        float val = acc[v] + bv;
        if (n == tgt[v]) ltg[v] = val;
        if (val > lmax[v]) { lsum[v] = lsum[v] * __expf(lmax[v] - val) + 1.f; lmax[v] = val; }
        else lsum[v] += __expf(val - lmax[v]);
      }
      acc = vzero;
    }
    __syncthreads();  // all waves done reading Bs[buf] before it is reloaded
  }

#pragma unroll
  for (int v = 0; v < 8; ++v) {
    float gm = lmax[v];
#pragma unroll
    for (int o = 8; o > 0; o >>= 1) gm = fmaxf(gm, __shfl_xor(gm, o, 32));
    float e = lsum[v] * __expf(lmax[v] - gm);
    float tg = ltg[v];
#pragma unroll
    for (int o = 8; o > 0; o >>= 1) { e += __shfl_xor(e, o, 32); tg += __shfl_xor(tg, o, 32); }
    if (nn == 0) {
      int row = r0 + v + 8 * half;
      Pmax[row * 16 + slice] = gm;
      Psum[row * 16 + slice] = e;
      Ptgt[row * 16 + slice] = tg;
    }
  }
}

__global__ void k_final(const float* __restrict__ Pmax, const float* __restrict__ Psum,
                        const float* __restrict__ Ptgt, const int* __restrict__ tgt_tokens,
                        float* __restrict__ out) {
  __shared__ float red[256];
  float local = 0.f;
  for (int row = threadIdx.x; row < cB * cT; row += 256) {
    float gm = -3e38f;
    for (int s = 0; s < 16; ++s) gm = fmaxf(gm, Pmax[row * 16 + s]);
    float gs = 0.f, tg = 0.f;
    for (int s = 0; s < 16; ++s) {
      gs += Psum[row * 16 + s] * __expf(Pmax[row * 16 + s] - gm);
      tg += Ptgt[row * 16 + s];
    }
    int b = row / cT, t = row % cT;
    int tok = tgt_tokens[b * (cT + 1) + t + 1];
    if (tok != 0) local += (gm + __logf(gs)) - tg;
  }
  red[threadIdx.x] = local;
  __syncthreads();
  for (int o = 128; o > 0; o >>= 1) {
    if (threadIdx.x < (unsigned)o) red[threadIdx.x] += red[threadIdx.x + o];
    __syncthreads();
  }
  if (threadIdx.x == 0) out[0] = red[0];
}

// ---------------- host ----------------
extern "C" void kernel_launch(void* const* d_in, const int* in_sizes, int n_in,
                              void* d_out, int out_size, void* d_ws, size_t ws_size,
                              hipStream_t stream) {
  (void)in_sizes; (void)n_in; (void)out_size; (void)ws_size;
  const float* enc_embed = (const float*)d_in[0];
  const float* dec_embed = (const float*)d_in[1];
  const float* eWihF = (const float*)d_in[2];
  const float* eWhhF = (const float*)d_in[3];
  const float* ebihF = (const float*)d_in[4];
  const float* ebhhF = (const float*)d_in[5];
  const float* eWihB = (const float*)d_in[6];
  const float* eWhhB = (const float*)d_in[7];
  const float* ebihB = (const float*)d_in[8];
  const float* ebhhB = (const float*)d_in[9];
  const float* dWih = (const float*)d_in[10];
  const float* dWhh = (const float*)d_in[11];
  const float* dbih = (const float*)d_in[12];
  const float* dbhh = (const float*)d_in[13];
  const float* WsW = (const float*)d_in[14];
  const float* Wsb = (const float*)d_in[15];
  const int* src_tokens = (const int*)d_in[16];
  const int* src_lengths = (const int*)d_in[17];
  const int* tgt_tokens = (const int*)d_in[18];

  char* w = (char*)d_ws;
  size_t off = 0;
  auto alloc = [&](size_t bytes) -> void* {
    void* p = w + off;
    off += (bytes + 255) & ~(size_t)255;
    return p;
  };

  unsigned* ctrl = (unsigned*)alloc(256);
  bf16_t* embF = (bf16_t*)alloc((size_t)cB * cS * cE * 2);
  bf16_t* embR = (bf16_t*)alloc((size_t)cB * cS * cE * 2);
  bf16_t* demb = (bf16_t*)alloc((size_t)cB * cT * cE * 2);
  bf16_t* WihFb = (bf16_t*)alloc((size_t)4 * cHH * cE * 2);
  bf16_t* WhhFb = (bf16_t*)alloc((size_t)4 * cHH * cHH * 2);
  bf16_t* WihBb = (bf16_t*)alloc((size_t)4 * cHH * cE * 2);
  bf16_t* WhhBb = (bf16_t*)alloc((size_t)4 * cHH * cHH * 2);
  bf16_t* Wdihb = (bf16_t*)alloc((size_t)4 * cH * (cE + cH) * 2);
  bf16_t* Wdhhb = (bf16_t*)alloc((size_t)4 * cH * cH * 2);
  bf16_t* Wsb16 = (bf16_t*)alloc((size_t)cV * 2 * cH * 2);
  bf16_t* hFbf = (bf16_t*)alloc((size_t)2 * cB * cHH * 2);
  bf16_t* hBbf = (bf16_t*)alloc((size_t)2 * cB * cHH * 2);
  bf16_t* habf = (bf16_t*)alloc((size_t)2 * cB * cH * 2);
  bf16_t* wabf = (bf16_t*)alloc((size_t)2 * cB * cH * 2);
  bf16_t* OUTS = (bf16_t*)alloc((size_t)cB * cT * 2 * cH * 2);
  float* XF = (float*)alloc((size_t)cB * cS * 4 * cHH * 4);
  float* XB = (float*)alloc((size_t)cB * cS * 4 * cHH * 4);
  float* PDE = (float*)alloc((size_t)cB * cT * 4 * cH * 4);
  float* outF = (float*)alloc((size_t)cB * cS * cHH * 4);
  float* outR = (float*)alloc((size_t)cB * cS * cHH * 4);
  float* hF = (float*)alloc((size_t)cB * cHH * 4);
  float* cF = (float*)alloc((size_t)cB * cHH * 4);
  float* hB = (float*)alloc((size_t)cB * cHH * 4);
  float* cB2 = (float*)alloc((size_t)cB * cHH * 4);
  float* hd = (float*)alloc((size_t)cB * cH * 4);
  float* cd = (float*)alloc((size_t)cB * cH * 4);
  float* senc = (float*)alloc((size_t)cB * cS * cH * 4);
  float* Pmax = (float*)alloc((size_t)cB * cT * 16 * 4);
  float* Psum = (float*)alloc((size_t)cB * cT * 16 * 4);
  float* Ptgt = (float*)alloc((size_t)cB * cT * 16 * 4);

  hipMemsetAsync(ctrl, 0, 256, stream);

  // weight conversions fp32 -> bf16
  k_cvt_bf16<<<512, 256, 0, stream>>>(eWihF, WihFb, (long)4 * cHH * cE);
  k_cvt_bf16<<<512, 256, 0, stream>>>(eWhhF, WhhFb, (long)4 * cHH * cHH);
  k_cvt_bf16<<<512, 256, 0, stream>>>(eWihB, WihBb, (long)4 * cHH * cE);
  k_cvt_bf16<<<512, 256, 0, stream>>>(eWhhB, WhhBb, (long)4 * cHH * cHH);
  k_cvt_bf16<<<1024, 256, 0, stream>>>(dWih, Wdihb, (long)4 * cH * (cE + cH));
  k_cvt_bf16<<<1024, 256, 0, stream>>>(dWhh, Wdhhb, (long)4 * cH * cH);
  k_cvt_bf16<<<2048, 256, 0, stream>>>(WsW, Wsb16, (long)cV * 2 * cH);

  // embeddings
  k_embed_enc<<<(cB * cS * cE + 255) / 256, 256, 0, stream>>>(
      enc_embed, src_tokens, src_lengths, embF, embR);
  k_embed_dec<<<(cB * cT * cE + 255) / 256, 256, 0, stream>>>(dec_embed, tgt_tokens, demb);

  // batched x-projections
  k_gemm_bias<<<768, 256, 0, stream>>>(embF, WihFb, cE, ebihF, ebhhF, XF,
                                       cB * cS, 4 * cHH, cE);
  k_gemm_bias<<<768, 256, 0, stream>>>(embR, WihBb, cE, ebihB, ebhhB, XB,
                                       cB * cS, 4 * cHH, cE);
  k_gemm_bias<<<768, 256, 0, stream>>>(demb, Wdihb + cH, cE + cH, dbih, dbhh, PDE,
                                       cB * cT, 4 * cH, cE);

  // encoder (fwd+bwd in one persistent kernel)
  k_enc_lstm<<<16, 256, 0, stream>>>(XF, XB, WhhFb, WhhBb, src_lengths,
                                     hFbf, hBbf, hF, cF, hB, cB2, outF, outR,
                                     ctrl, ctrl + 1);

  // src_enc + decoder init state
  k_postenc<<<(cB * cS * cH + 255) / 256, 256, 0, stream>>>(outF, outR, src_lengths, senc);
  k_initdec<<<(cB * cH + 255) / 256, 256, 0, stream>>>(hF, hB, cF, cB2, hd, cd,
                                                       habf, wabf);

  // decoder with attention (persistent)
  k_dec_lstm<<<16, 256, 0, stream>>>(PDE, Wdihb, Wdhhb, senc, src_lengths,
                                     habf, wabf, hd, cd, OUTS, ctrl + 2, ctrl + 3);

  // fused vocab projection + online softmax partials (TDM-staged B tiles)
  k_vocab<<<192, 256, 0, stream>>>(OUTS, Wsb16, Wsb, tgt_tokens, Pmax, Psum, Ptgt);

  // final NLL reduction
  k_final<<<1, 256, 0, stream>>>(Pmax, Psum, Ptgt, tgt_tokens, (float*)d_out);
}
